// SpatialAttention_48954037240015
// MI455X (gfx1250) — matrix operand order repaired and checked by reading, not compiled, not run on hardware
//
#include <hip/hip_runtime.h>
#include <hip/hip_bf16.h>

typedef __attribute__((ext_vector_type(16))) __bf16 v16bf;
typedef __attribute__((ext_vector_type(8)))  float  v8f;

#define DEV static __device__ __forceinline__

constexpr int NB = 8, NT = 12, NN = 512, HD = 32, NH = 4, ND = 128, C2 = 256, QC = 384;
constexpr int MROWS = NB * NT * NN;          // 49152 flattened activation rows
constexpr float EPSBN = 1e-5f;
constexpr float NEGF  = -32767.0f;           // -2^15 + 1 mask fill

// ---------- bf16 helpers ----------
// native fptrunc -> v_cvt (RNE) instead of the 3-op integer sequence
DEV unsigned short f2bfbits(float f) {
  __bf16 h = (__bf16)f;
  return __builtin_bit_cast(unsigned short, h);
}

// build a 16-element bf16 fragment from two 16-byte chunks (must be 16B aligned)
struct B32c { uint4 lo, hi; };
DEV v16bf frag2(const unsigned short* p0, const unsigned short* p1) {
  B32c t;
  t.lo = *(const uint4*)p0;
  t.hi = *(const uint4*)p1;
  return __builtin_bit_cast(v16bf, t);
}

DEV v8f wmma_bf16(v16bf a, v16bf b, v8f c) {
  return __builtin_amdgcn_wmma_f32_16x16x32_bf16(false, a, false, b, (short)0, c, false, false);
}

// reductions across the 16 lanes sharing a matrix row (xor masks 1..8 stay in the group)
DEV float redmax16(float v) {
  #pragma unroll
  for (int m = 1; m < 16; m <<= 1) v = fmaxf(v, __shfl_xor(v, m, 32));
  return v;
}
DEV float redsum16(float v) {
  #pragma unroll
  for (int m = 1; m < 16; m <<= 1) v += __shfl_xor(v, m, 32);
  return v;
}

// ---------- kernel 0: zero BN accumulators ----------
__global__ void zero_sums_kernel(float* p, int n) {
  int i = blockIdx.x * blockDim.x + threadIdx.x;
  if (i < n) p[i] = 0.0f;
}

// ---------- conversion kernels (one-time, bandwidth-bound) ----------
// concat(X,STE) -> bf16 [49152 x 256], 4 elements per thread
__global__ void convert_cat_kernel(const float* __restrict__ X, const float* __restrict__ STE,
                                   unsigned short* __restrict__ catbf) {
  const int i = blockIdx.x * blockDim.x + threadIdx.x;      // quad index
  const int flat = i * 4;
  const int row = flat >> 8, c = flat & 255;
  const float* src = (c < ND) ? (X + (size_t)row * ND + c) : (STE + (size_t)row * ND + (c - ND));
  float4 u = *(const float4*)src;
  uint2 o;
  o.x = (unsigned)f2bfbits(u.x) | ((unsigned)f2bfbits(u.y) << 16);
  o.y = (unsigned)f2bfbits(u.z) | ((unsigned)f2bfbits(u.w) << 16);
  *(uint2*)(catbf + flat) = o;
}

// [Wq|Wk|Wv] (256 x 384, row-major) -> transposed bf16 WqkvT [384 x 256]
__global__ void convert_wqkv_kernel(const float* __restrict__ Wq, const float* __restrict__ Wk,
                                    const float* __restrict__ Wv, unsigned short* __restrict__ WqkvT) {
  const int idx = blockIdx.x * blockDim.x + threadIdx.x;    // 384*256
  const int n = idx >> 8, k = idx & 255;
  const int sec = n >> 7;
  const float* W = (sec == 0) ? Wq : (sec == 1) ? Wk : Wv;
  WqkvT[(size_t)n * C2 + k] = f2bfbits(W[(size_t)k * ND + (n & (ND - 1))]);
}

// Wo (128 x 128) -> transposed bf16 WoT [128 x 128]
__global__ void convert_wo_kernel(const float* __restrict__ Wo, unsigned short* __restrict__ WoT) {
  const int idx = blockIdx.x * blockDim.x + threadIdx.x;    // 128*128
  const int n = idx >> 7, k = idx & 127;
  WoT[(size_t)n * ND + k] = f2bfbits(Wo[(size_t)k * ND + n]);
}

// ---------- kernel 1: fused QKV GEMM (bf16 WMMA, fp32 accum) ----------
__global__ void qkv_gemm_kernel(const unsigned short* __restrict__ catbf,
                                const unsigned short* __restrict__ WqkvT,
                                const float* __restrict__ bq, const float* __restrict__ bk,
                                const float* __restrict__ bv,
                                float* __restrict__ ypre,
                                float* __restrict__ sum, float* __restrict__ sumsq) {
  const int lane = threadIdx.x & 31, wv = threadIdx.x >> 5;
  const int l16 = lane & 15, hi = lane >> 4;
  const int wg = blockIdx.x * 4 + wv;         // one 16x16 tile per wave
  const int tm = wg / (QC / 16);              // 0..3071
  const int tn = wg % (QC / 16);              // 0..23
  const int gcol = tn * 16 + l16;             // 0..383
  const int row  = tm * 16 + l16;

  const unsigned short* arow = catbf + (size_t)row * C2;
  const unsigned short* bcol = WqkvT + (size_t)gcol * C2;

  v8f acc = {};
  #pragma unroll
  for (int kk = 0; kk < C2; kk += 32) {
    const unsigned short* ap = arow + kk + hi * 8;    // A: chunks at +0 and +16
    const unsigned short* bp = bcol + kk + hi * 8;    // B: chunks at +0 and +16
    if (kk + 32 < C2) __builtin_prefetch(bp + 32, 0, 1);
    v16bf a = frag2(ap, ap + 16);
    v16bf b = frag2(bp, bp + 16);
    acc = wmma_bf16(a, b, acc);
  }

  const int sec = gcol >> 7;
  const float* bias = (sec == 0) ? bq : (sec == 1) ? bk : bv;
  const float bb = bias[gcol & (ND - 1)];
  float ls = 0.f, lq = 0.f;
  #pragma unroll
  for (int r = 0; r < 8; ++r) {
    const int grow = tm * 16 + hi * 8 + r;
    float v = acc[r] + bb;
    ypre[(size_t)grow * QC + gcol] = v;
    ls += v; lq += v * v;
  }
  ls += __shfl_xor(ls, 16, 32);               // lanes l and l+16 share the same column
  lq += __shfl_xor(lq, 16, 32);
  if (hi == 0) { atomicAdd(&sum[gcol], ls); atomicAdd(&sumsq[gcol], lq); }
}

// ---------- kernel 2: BN stats finalize ----------
__global__ void bn_stats_kernel(const float* __restrict__ sum, const float* __restrict__ sumsq,
                                float* __restrict__ stats, int C, float inv_cnt) {
  int c = threadIdx.x;
  if (c < C) {
    float mu  = sum[c] * inv_cnt;
    float var = sumsq[c] * inv_cnt - mu * mu;
    stats[c]     = mu;
    stats[C + c] = rsqrtf(var + EPSBN);
  }
}

// ---------- kernel 3: masked multi-head attention (flash-style, all WMMA) ----------
// 1 block per (h, b*T+t); 512 threads = 16 waves; each wave owns two 16-row tiles.
__global__ void attn_kernel(const float* __restrict__ ypre, const float* __restrict__ stats,
                            const float* __restrict__ gq, const float* __restrict__ betaq,
                            const float* __restrict__ gk, const float* __restrict__ betak,
                            const float* __restrict__ gv, const float* __restrict__ betav,
                            const float* __restrict__ Adj, unsigned short* __restrict__ attn_out) {
  extern __shared__ unsigned short smem[];
  unsigned short* qs  = smem;                 // [512][32] bf16, row-major
  unsigned short* ks_ = qs  + NN * HD;        // [512][32]
  unsigned short* vsT = ks_ + NN * HD;        // [32][512] (transposed for contiguous PV B-frags)
  unsigned short* ps  = vsT + NN * HD;        // 16 waves * [16][32] P scratch

  const int p  = blockIdx.x;
  const int h  = p / (NB * NT);
  const int bt = p % (NB * NT);
  const int tid = threadIdx.x;

  // stage q/k/v into LDS with BN (batch stats) + ReLU, converted to bf16
  for (int f = tid; f < NN * HD; f += 512) {
    const int n = f >> 5, c = f & 31;
    const size_t grow = (size_t)(bt * NN + n);
    const int ch = h * HD + c;
    { const int col = ch;                     // q section
      float y = ypre[grow * QC + col];
      float v = (y - stats[col]) * stats[QC + col] * gq[ch] + betaq[ch];
      qs[f] = f2bfbits(fmaxf(v, 0.f)); }
    { const int col = ND + ch;                // k section
      float y = ypre[grow * QC + col];
      float v = (y - stats[col]) * stats[QC + col] * gk[ch] + betak[ch];
      ks_[f] = f2bfbits(fmaxf(v, 0.f)); }
    { const int col = 2 * ND + ch;            // v section (store transposed)
      float y = ypre[grow * QC + col];
      float v = (y - stats[col]) * stats[QC + col] * gv[ch] + betav[ch];
      vsT[c * NN + n] = f2bfbits(fmaxf(v, 0.f)); }
  }
  __syncthreads();

  const int lane = tid & 31, wv = tid >> 5;
  const int l16 = lane & 15, hi = lane >> 4;
  const int kb8 = hi * 8;

  for (int it = 0; it < 2; ++it) {
    const int rt = it * 16 + wv;              // row tile 0..31
    const int rowb = rt * 16;

    const unsigned short* qr = qs + (rowb + l16) * HD;
    v16bf aq = frag2(qr + kb8, qr + kb8 + 16);  // Q fragment (16 rows x K=32)

    float m8[8], l8[8];
    #pragma unroll
    for (int r = 0; r < 8; ++r) { m8[r] = -3.0e38f; l8[r] = 0.f; }
    v8f acc0 = {}, acc1 = {};

    for (int jb = 0; jb < NN; jb += 32) {     // 32 keys per step
      const unsigned short* kc0 = ks_ + (jb + l16) * HD + kb8;
      const unsigned short* kc1 = ks_ + (jb + 16 + l16) * HD + kb8;
      v16bf bk0 = frag2(kc0, kc0 + 16);
      v16bf bk1 = frag2(kc1, kc1 + 16);
      v8f z = {};
      v8f s0 = wmma_bf16(aq, bk0, z);
      v8f s1 = wmma_bf16(aq, bk1, z);

      #pragma unroll
      for (int r = 0; r < 8; ++r) {
        const int nrow = rowb + hi * 8 + r;
        const float a0 = Adj[(size_t)nrow * NN + jb + l16];
        const float a1 = Adj[(size_t)nrow * NN + jb + 16 + l16];
        float v0 = (a0 > 0.f) ? s0[r] * 0.5f : NEGF;   // scale 1/sqrt(d)=0.5, then mask
        float v1 = (a1 > 0.f) ? s1[r] * 0.5f : NEGF;
        const float mx = redmax16(fmaxf(v0, v1));
        const float mn = fmaxf(m8[r], mx);
        const float corr = __expf(m8[r] - mn);
        m8[r] = mn;
        const float p0 = __expf(v0 - mn), p1 = __expf(v1 - mn);
        l8[r] = l8[r] * corr + redsum16(p0 + p1);
        acc0[r] *= corr; acc1[r] *= corr;
        const int rr = hi * 8 + r;            // re-layout P (D-layout) -> LDS for A-fragment
        ps[(wv * 16 + rr) * 32 + l16]      = f2bfbits(p0);
        ps[(wv * 16 + rr) * 32 + 16 + l16] = f2bfbits(p1);
      }
      asm volatile("s_wait_dscnt 0" ::: "memory");      // per-wave LDS RAW fence

      const unsigned short* pr = ps + (wv * 16 + l16) * 32;
      v16bf ap = frag2(pr + kb8, pr + kb8 + 16);
      const unsigned short* vb0 = vsT + (size_t)l16 * NN + jb + kb8;
      const unsigned short* vb1 = vsT + (size_t)(16 + l16) * NN + jb + kb8;
      v16bf bv0 = frag2(vb0, vb0 + 16);
      v16bf bv1 = frag2(vb1, vb1 + 16);
      acc0 = wmma_bf16(ap, bv0, acc0);
      acc1 = wmma_bf16(ap, bv1, acc1);
    }

    #pragma unroll
    for (int r = 0; r < 8; ++r) {
      const size_t grow = (size_t)(bt * NN + rowb + hi * 8 + r);
      const float inv = 1.0f / l8[r];
      attn_out[grow * ND + h * HD + l16]      = f2bfbits(acc0[r] * inv);
      attn_out[grow * ND + h * HD + 16 + l16] = f2bfbits(acc1[r] * inv);
    }
  }
}

// ---------- kernel 4: output projection GEMM (bf16 WMMA) ----------
__global__ void out_gemm_kernel(const unsigned short* __restrict__ Ain,
                                const unsigned short* __restrict__ WoT,
                                const float* __restrict__ bo,
                                float* __restrict__ ypre,
                                float* __restrict__ sum, float* __restrict__ sumsq) {
  const int lane = threadIdx.x & 31, wv = threadIdx.x >> 5;
  const int l16 = lane & 15, hi = lane >> 4;
  const int wg = blockIdx.x * 4 + wv;
  const int tm = wg / (ND / 16);
  const int tn = wg % (ND / 16);
  const int coln = tn * 16 + l16;
  const int row  = tm * 16 + l16;

  const unsigned short* arow = Ain + (size_t)row * ND;
  const unsigned short* bcol = WoT + (size_t)coln * ND;

  v8f acc = {};
  #pragma unroll
  for (int kk = 0; kk < ND; kk += 32) {
    const unsigned short* ap = arow + kk + hi * 8;
    const unsigned short* bp = bcol + kk + hi * 8;
    v16bf a = frag2(ap, ap + 16);
    v16bf b = frag2(bp, bp + 16);
    acc = wmma_bf16(a, b, acc);
  }
  const float bb = bo[coln];
  float ls = 0.f, lq = 0.f;
  #pragma unroll
  for (int r = 0; r < 8; ++r) {
    const int grow = tm * 16 + hi * 8 + r;
    float v = acc[r] + bb;
    ypre[(size_t)grow * ND + coln] = v;
    ls += v; lq += v * v;
  }
  ls += __shfl_xor(ls, 16, 32);
  lq += __shfl_xor(lq, 16, 32);
  if (hi == 0) { atomicAdd(&sum[coln], ls); atomicAdd(&sumsq[coln], lq); }
}

// ---------- kernel 6: final BN + ReLU ----------
__global__ void bn_apply_kernel(const float* __restrict__ ypre, const float* __restrict__ stats,
                                const float* __restrict__ g, const float* __restrict__ beta,
                                float* __restrict__ out, int total) {
  int i = blockIdx.x * blockDim.x + threadIdx.x;
  if (i < total) {
    const int c = i & (ND - 1);
    float v = (ypre[i] - stats[c]) * stats[ND + c] * g[c] + beta[c];
    out[i] = fmaxf(v, 0.f);
  }
}

extern "C" void kernel_launch(void* const* d_in, const int* in_sizes, int n_in,
                              void* d_out, int out_size, void* d_ws, size_t ws_size,
                              hipStream_t stream) {
  const float* X    = (const float*)d_in[0];
  const float* STE  = (const float*)d_in[1];
  const float* Adj  = (const float*)d_in[2];
  const float* Wq   = (const float*)d_in[3];
  const float* bq   = (const float*)d_in[4];
  const float* gq   = (const float*)d_in[5];
  const float* betaq= (const float*)d_in[6];
  const float* Wk   = (const float*)d_in[7];
  const float* bk   = (const float*)d_in[8];
  const float* gk   = (const float*)d_in[9];
  const float* betak= (const float*)d_in[10];
  const float* Wv   = (const float*)d_in[11];
  const float* bv   = (const float*)d_in[12];
  const float* gv   = (const float*)d_in[13];
  const float* betav= (const float*)d_in[14];
  const float* Wo   = (const float*)d_in[15];
  const float* bo   = (const float*)d_in[16];
  const float* go   = (const float*)d_in[17];
  const float* betao= (const float*)d_in[18];

  // workspace carve-up (all segments 16B aligned)
  float* ws        = (float*)d_ws;
  float* qkv_pre   = ws;                                     // 49152*384 fp32
  float* sums      = qkv_pre + (size_t)MROWS * QC;           // 1024 fp32
  float* sum_qkv   = sums;
  float* sumsq_qkv = sums + QC;
  float* sum_o     = sums + 2 * QC;
  float* sumsq_o   = sums + 2 * QC + ND;
  float* stats_qkv = sums + 1024;                            // 2*384
  float* stats_o   = stats_qkv + 2 * QC;                     // 2*128
  float* out_pre   = stats_o + 2 * ND;                       // 49152*128 fp32
  unsigned short* attn_out = (unsigned short*)(out_pre + (size_t)MROWS * ND); // 49152*128 bf16
  unsigned short* catbf    = attn_out + (size_t)MROWS * ND;  // 49152*256 bf16
  unsigned short* WqkvT    = catbf + (size_t)MROWS * C2;     // 384*256 bf16 (transposed)
  unsigned short* WoT      = WqkvT + (size_t)QC * C2;        // 128*128 bf16 (transposed)

  zero_sums_kernel<<<1, 1024, 0, stream>>>(sums, 1024);

  convert_cat_kernel<<<(MROWS * C2 / 4) / 256, 256, 0, stream>>>(X, STE, catbf);
  convert_wqkv_kernel<<<(QC * C2) / 256, 256, 0, stream>>>(Wq, Wk, Wv, WqkvT);
  convert_wo_kernel<<<(ND * ND) / 256, 256, 0, stream>>>(Wo, WoT);

  // QKV GEMM: 3072*24 tiles, 4 waves/block
  qkv_gemm_kernel<<<(MROWS / 16) * (QC / 16) / 4, 128, 0, stream>>>(
      catbf, WqkvT, bq, bk, bv, qkv_pre, sum_qkv, sumsq_qkv);

  bn_stats_kernel<<<1, QC, 0, stream>>>(sum_qkv, sumsq_qkv, stats_qkv, QC, 1.0f / (float)MROWS);

  const size_t shmem = (size_t)(3 * NN * HD + 16 * 16 * 32) * sizeof(unsigned short); // 112 KB
  attn_kernel<<<NH * NB * NT, 512, shmem, stream>>>(
      qkv_pre, stats_qkv, gq, betaq, gk, betak, gv, betav, Adj, attn_out);

  out_gemm_kernel<<<(MROWS / 16) * (ND / 16) / 4, 128, 0, stream>>>(
      attn_out, WoT, bo, out_pre, sum_o, sumsq_o);

  bn_stats_kernel<<<1, ND, 0, stream>>>(sum_o, sumsq_o, stats_o, ND, 1.0f / (float)MROWS);

  const int total = MROWS * ND;
  bn_apply_kernel<<<(total + 255) / 256, 256, 0, stream>>>(out_pre, stats_o, go, betao,
                                                           (float*)d_out, total);
}
